// GQA_66108136620702
// MI455X (gfx1250) — matrix operand order repaired and checked by reading, not compiled, not run on hardware
//
#include <hip/hip_runtime.h>
#include <hip/hip_bf16.h>
#include <cstdint>
#include <cstddef>

// ---------------------------------------------------------------- constants
#define BB   2
#define TT   2048
#define DD   1024
#define HH   16
#define KVH  4
#define DH   64
#define WIN  512

typedef __attribute__((ext_vector_type(16))) __bf16 v16bf;
typedef __attribute__((ext_vector_type(8)))  float  v8f;
typedef __attribute__((ext_vector_type(2)))  __bf16 bf16x2;
typedef __attribute__((ext_vector_type(4)))  __bf16 bf16x4;
typedef __attribute__((ext_vector_type(4)))  unsigned int u32x4;
typedef __attribute__((ext_vector_type(8)))  int i32x8;
typedef __attribute__((ext_vector_type(4)))  int i32x4;

#if defined(__has_builtin)
#  if __has_builtin(__builtin_amdgcn_tensor_load_to_lds) && __has_builtin(__builtin_amdgcn_s_wait_tensorcnt)
#    define HAVE_TDM 1
#  endif
#endif
#ifndef HAVE_TDM
#  define HAVE_TDM 0
#endif

static __device__ __forceinline__ __bf16 f2bf(float f) {
    unsigned u = __builtin_bit_cast(unsigned, f);
    u += 0x7fffu + ((u >> 16) & 1u);               // round-to-nearest-even
    unsigned short h = (unsigned short)(u >> 16);
    return __builtin_bit_cast(__bf16, h);
}
static __device__ __forceinline__ bf16x2 ld_bf2(const __bf16* p) {
    return *reinterpret_cast<const bf16x2*>(p);
}
static __device__ __forceinline__ v8f wmma_bf16(v16bf a, v16bf b, v8f c) {
    return __builtin_amdgcn_wmma_f32_16x16x32_bf16(false, a, false, b,
                                                   (short)0, c, false, false);
}

#if HAVE_TDM
// 2D TDM load: tile tile_d1 rows x tile_d0(=32) bf16, row stride row_stride
// elems; LDS padding: +2 DWORDs per 64B row => LDS row stride = 36 bf16 (72B,
// 8B-aligned so ds_load_2addr_b64 fusing works).
static __device__ __forceinline__ void tdm_load_2d(unsigned lds_off,
        const __bf16* gptr, int tile_d0, int tile_d1, int row_stride_elems)
{
    unsigned long long ga = (unsigned long long)(uintptr_t)gptr;
    u32x4 g0;
    g0[0] = 1u;                                   // count=1 (valid), user mode
    g0[1] = lds_off;                              // LDS byte address
    g0[2] = (unsigned)ga;                         // global_addr[31:0]
    g0[3] = (unsigned)((ga >> 32) & 0x01ffffffull) | (2u << 30);  // addr[56:32] | type=2
    i32x8 g1;
    // data_size=2B, pad_enable, pad_interval=16 DWORDs (64B), pad_amount=2 DWORDs
    g1[0] = (1 << 16) | (1 << 20) | (3 << 22) | (1 << 25);
    g1[1] = (int)((unsigned)tile_d0 << 16);                               // tensor_dim0 lo
    g1[2] = (int)(((unsigned)tile_d0 >> 16) | ((unsigned)tile_d1 << 16)); // td0 hi | td1 lo
    g1[3] = (int)(((unsigned)tile_d1 >> 16) | ((unsigned)tile_d0 << 16)); // td1 hi | tile_dim0
    g1[4] = tile_d1 & 0xffff;                                             // tile_dim1
    g1[5] = row_stride_elems;                                             // dim0_stride lo
    g1[6] = 0;
    g1[7] = 0;
    i32x4 z4 = {0, 0, 0, 0};
#if defined(__clang_major__) && (__clang_major__ >= 23)
    i32x8 z8 = {0, 0, 0, 0, 0, 0, 0, 0};
    __builtin_amdgcn_tensor_load_to_lds(g0, g1, z4, z4, z8, 0);
#else
    __builtin_amdgcn_tensor_load_to_lds(g0, g1, z4, z4, 0);
#endif
}
#endif

// ---------------------------------------------------------------- f32 -> bf16
__global__ void __launch_bounds__(256)
cvt_f32_bf16(const float* __restrict__ src, __bf16* __restrict__ dst)
{
    int i = (blockIdx.x * 256 + threadIdx.x) * 4;
    float4 v = *reinterpret_cast<const float4*>(src + i);
    bf16x4 o;
    o[0] = f2bf(v.x); o[1] = f2bf(v.y); o[2] = f2bf(v.z); o[3] = f2bf(v.w);
    *reinterpret_cast<bf16x4*>(dst + i) = o;
}

// ---------------------------------------------------------------- GEMM
// C[M,Ntot] = A[M,K] * W^T, bf16 in, f32 out.
// block: 256 threads (8 waves), tile 128(M) x 128(N), K-step 32, TDM-staged
// double-buffered LDS; wave tile 32x64 => 8 WMMAs per wave per K-step.
__global__ void __launch_bounds__(256)
gemm_bf16_wmma(const __bf16* __restrict__ A,
               const __bf16* __restrict__ W,
               float* __restrict__ C,
               int K, int Ntot)
{
    __shared__ __align__(16) __bf16 As[2][128][36];
    __shared__ __align__(16) __bf16 Bs[2][128][36];

    const int tid  = threadIdx.x;
    const int lane = tid & 31;
    const int wave = tid >> 5;
    const int wm   = (wave & 3) << 5;      // 0,32,64,96
    const int wn   = (wave >> 2) << 6;     // 0,64
    const int m0g  = blockIdx.x * 128;
    const int n0g  = blockIdx.y * 128;
    const __bf16* Abase = A + (size_t)m0g * K;
    const __bf16* Wbase = W + (size_t)n0g * K;

    v8f acc[2][4] = {};
    int cur = 0;

#if HAVE_TDM
    if (wave == 0) {
        tdm_load_2d((unsigned)(uintptr_t)&As[0][0][0], Abase, 32, 128, K);
        tdm_load_2d((unsigned)(uintptr_t)&Bs[0][0][0], Wbase, 32, 128, K);
    }
#endif

    for (int k0 = 0; k0 < K; k0 += 32) {
#if HAVE_TDM
        if (wave == 0) {
            if (k0 + 32 < K) {
                tdm_load_2d((unsigned)(uintptr_t)&As[cur ^ 1][0][0], Abase + k0 + 32, 32, 128, K);
                tdm_load_2d((unsigned)(uintptr_t)&Bs[cur ^ 1][0][0], Wbase + k0 + 32, 32, 128, K);
                __builtin_amdgcn_s_wait_tensorcnt(2);   // current stage done
            } else {
                __builtin_amdgcn_s_wait_tensorcnt(0);
            }
        }
        __syncthreads();
#else
        // cooperative fallback: two 128x32 bf16 tiles
        #pragma unroll
        for (int it = 0; it < 2; ++it) {
            int idx = tid + it * 256;                 // 0..511
            int row = idx >> 2;
            int col = (idx & 3) << 3;
            const __bf16* ap = Abase + (size_t)row * K + k0 + col;
            const __bf16* wp = Wbase + (size_t)row * K + k0 + col;
            union { uint4 u; __bf16 h[8]; } ta, tw;
            ta.u = *reinterpret_cast<const uint4*>(ap);
            tw.u = *reinterpret_cast<const uint4*>(wp);
            #pragma unroll
            for (int e = 0; e < 8; ++e) {
                As[cur][row][col + e] = ta.h[e];
                Bs[cur][row][col + e] = tw.h[e];
            }
            if (k0 + 32 < K) {
                __builtin_prefetch(ap + 32, 0, 3);
                __builtin_prefetch(wp + 32, 0, 3);
            }
        }
        __syncthreads();
#endif

        // ---- fragments + WMMA
        const int ha = lane >> 4;
        v16bf af[2], bfr[4];
        #pragma unroll
        for (int mt = 0; mt < 2; ++mt) {
            int row = wm + mt * 16 + (lane & 15);
            #pragma unroll
            for (int vv = 0; vv < 8; ++vv) {
                int kk = ((vv & 3) << 1) + (ha << 3) + ((vv >> 2) << 4);
                bf16x2 p = ld_bf2(&As[cur][row][kk]);
                af[mt][2 * vv] = p[0]; af[mt][2 * vv + 1] = p[1];
            }
        }
        #pragma unroll
        for (int nt = 0; nt < 4; ++nt) {
            int row = wn + nt * 16 + (lane & 15);
            #pragma unroll
            for (int vv = 0; vv < 8; ++vv) {
                int kk = ((vv & 3) << 1) + (ha << 3) + ((vv >> 2) << 4);
                bf16x2 p = ld_bf2(&Bs[cur][row][kk]);
                bfr[nt][2 * vv] = p[0]; bfr[nt][2 * vv + 1] = p[1];
            }
        }
        #pragma unroll
        for (int mt = 0; mt < 2; ++mt)
            #pragma unroll
            for (int nt = 0; nt < 4; ++nt)
                acc[mt][nt] = wmma_bf16(af[mt], bfr[nt], acc[mt][nt]);
        __syncthreads();
        cur ^= 1;
    }

    // ---- store C (f32): vgpr r -> row r + 8*(lane/16), col = lane%16
    #pragma unroll
    for (int mt = 0; mt < 2; ++mt)
        #pragma unroll
        for (int nt = 0; nt < 4; ++nt)
            #pragma unroll
            for (int r = 0; r < 8; ++r) {
                int m = m0g + wm + mt * 16 + r + ((lane >> 4) << 3);
                int n = n0g + wn + nt * 16 + (lane & 15);
                C[(size_t)m * Ntot + n] = acc[mt][nt][r];
            }
}

// ---------------------------------------------------------------- RoPE / convert
__global__ void __launch_bounds__(256)
rope_q_kernel(const float* __restrict__ qkv, __bf16* __restrict__ Qbf)
{
    int tid = blockIdx.x * 256 + threadIdx.x;      // B*T*H*32 threads
    int i = tid & 31, h = (tid >> 5) & 15, t = (tid >> 9) & 2047, b = tid >> 20;
    const float* row = qkv + (size_t)(b * TT + t) * 1536 + h * DH;
    float x1 = row[2 * i], x2 = row[2 * i + 1];
    float invf = __expf(-(float)(2 * i) * (9.210340371976184f / 64.0f));
    float fr = (float)t * invf, s, c;
    __sincosf(fr, &s, &c);
    __bf16* dst = Qbf + (((size_t)b * HH + h) * TT + t) * DH;
    dst[2 * i]     = f2bf(x1 * c - x2 * s);
    dst[2 * i + 1] = f2bf(x1 * s + x2 * c);
}

__global__ void __launch_bounds__(256)
rope_k_kernel(const float* __restrict__ qkv, __bf16* __restrict__ Kbf)
{
    int tid = blockIdx.x * 256 + threadIdx.x;      // B*T*KV*32 threads
    int i = tid & 31, kv = (tid >> 5) & 3, t = (tid >> 7) & 2047, b = tid >> 18;
    const float* row = qkv + (size_t)(b * TT + t) * 1536 + 1024 + kv * DH;
    float x1 = row[2 * i], x2 = row[2 * i + 1];
    float invf = __expf(-(float)(2 * i) * (9.210340371976184f / 64.0f));
    float fr = (float)t * invf, s, c;
    __sincosf(fr, &s, &c);
    __bf16* dst = Kbf + (((size_t)b * KVH + kv) * TT + t) * DH;
    dst[2 * i]     = f2bf(x1 * c - x2 * s);
    dst[2 * i + 1] = f2bf(x1 * s + x2 * c);
}

__global__ void __launch_bounds__(256)
conv_v_kernel(const float* __restrict__ qkv, __bf16* __restrict__ Vbf)
{
    int tid = blockIdx.x * 256 + threadIdx.x;      // B*T*KV*64 threads
    int d = tid & 63, kv = (tid >> 6) & 3, t = (tid >> 8) & 2047, b = tid >> 19;
    float v = qkv[(size_t)(b * TT + t) * 1536 + 1280 + kv * DH + d];
    Vbf[(((size_t)b * KVH + kv) * DH + d) * TT + t] = f2bf(v);   // transposed [B,KV,Dh,T]
}

// ---------------------------------------------------------------- flash attention
// grid (T/64, H, B), 128 threads = 4 waves, wave handles a 16-query tile
__global__ void __launch_bounds__(128)
attn_kernel(const __bf16* __restrict__ Qbf,   // [B,H,T,Dh]
            const __bf16* __restrict__ Kbf,   // [B,KV,T,Dh]
            const __bf16* __restrict__ Vbf,   // [B,KV,Dh,T]
            __bf16* __restrict__ O)           // [B,T,H*Dh] bf16, feeds final GEMM
{
    __shared__ __align__(16) __bf16 Plds[4][16][36];

    const int b = blockIdx.z, h = blockIdx.y, kv = h >> 2;
    const int wave = threadIdx.x >> 5, lane = threadIdx.x & 31;
    const int qt0 = blockIdx.x * 64 + wave * 16;
    const int M = lane & 15, ha = lane >> 4;

    const __bf16* Qb = Qbf + (((size_t)b * HH + h) * TT) * DH;
    const __bf16* Kb = Kbf + (((size_t)b * KVH + kv) * TT) * DH;
    const __bf16* Vb = Vbf + (((size_t)b * KVH + kv) * DH) * (size_t)TT;

    // Q A-fragments (16 x 64 = two 16x32 k-chunks)
    v16bf qa[2];
    {
        const __bf16* qr = &Qb[(size_t)(qt0 + M) * DH];
        #pragma unroll
        for (int j = 0; j < 2; ++j)
            #pragma unroll
            for (int vv = 0; vv < 8; ++vv) {
                int d = ((vv & 3) << 1) + (ha << 3) + ((vv >> 2) << 4) + j * 32;
                bf16x2 p = ld_bf2(&qr[d]);
                qa[j][2 * vv] = p[0]; qa[j][2 * vv + 1] = p[1];
            }
    }

    float m_run[8], l_run[8];
    v8f o[4] = {};
    #pragma unroll
    for (int r = 0; r < 8; ++r) { m_run[r] = -1e30f; l_run[r] = 0.0f; }

    int kstart = qt0 - (WIN - 1); if (kstart < 0) kstart = 0;
    const int c0 = kstart & ~31;
    const int cend = qt0 + 16;                 // exclusive
    const float scale = 0.125f;                // 1/sqrt(64)

    for (int c = c0; c < cend; c += 32) {
        // near-cache prefetch of next key chunk (speculative; OOB dropped)
        __builtin_prefetch(&Kb[(size_t)(c + 32 + M) * DH], 0, 3);

        // ---- S = Q K^T  (two 16x16 key tiles)
        v8f s[2] = {};
        #pragma unroll
        for (int nt = 0; nt < 2; ++nt) {
            int key = c + nt * 16 + M;
            int keyc = key < (TT - 1) ? key : (TT - 1);
            const __bf16* kr = &Kb[(size_t)keyc * DH];
            #pragma unroll
            for (int dc = 0; dc < 2; ++dc) {
                v16bf kbf;
                #pragma unroll
                for (int vv = 0; vv < 8; ++vv) {
                    int d = ((vv & 3) << 1) + (ha << 3) + ((vv >> 2) << 4) + dc * 32;
                    bf16x2 p = ld_bf2(&kr[d]);
                    kbf[2 * vv] = p[0]; kbf[2 * vv + 1] = p[1];
                }
                s[nt] = wmma_bf16(qa[dc], kbf, s[nt]);
            }
        }

        // ---- mask + online softmax (row = r + 8*ha, cols across lanes)
        #pragma unroll
        for (int r = 0; r < 8; ++r) {
            int qi = qt0 + r + (ha << 3);
            int kc0 = c + M, kc1 = kc0 + 16;
            float x0 = (kc0 <= qi && kc0 >= qi - (WIN - 1)) ? s[0][r] * scale : -1e30f;
            float x1 = (kc1 <= qi && kc1 >= qi - (WIN - 1)) ? s[1][r] * scale : -1e30f;
            float mx = fmaxf(x0, x1);
            #pragma unroll
            for (int off = 1; off < 16; off <<= 1)
                mx = fmaxf(mx, __shfl_xor(mx, off, 32));
            float mnew  = fmaxf(m_run[r], mx);
            float alpha = __expf(m_run[r] - mnew);
            float p0 = __expf(x0 - mnew), p1 = __expf(x1 - mnew);
            float rs = p0 + p1;
            #pragma unroll
            for (int off = 1; off < 16; off <<= 1)
                rs += __shfl_xor(rs, off, 32);
            l_run[r] = l_run[r] * alpha + rs;
            m_run[r] = mnew;
            #pragma unroll
            for (int dt = 0; dt < 4; ++dt) o[dt][r] *= alpha;
            Plds[wave][r + (ha << 3)][M]      = f2bf(p0);
            Plds[wave][r + (ha << 3)][M + 16] = f2bf(p1);
        }

        // ---- P (C-layout -> A-layout via per-wave LDS slab)
        v16bf pa;
        #pragma unroll
        for (int vv = 0; vv < 8; ++vv) {
            int kk = ((vv & 3) << 1) + (ha << 3) + ((vv >> 2) << 4);
            bf16x2 p = ld_bf2(&Plds[wave][M][kk]);
            pa[2 * vv] = p[0]; pa[2 * vv + 1] = p[1];
        }

        // ---- O += P V  (4 dh tiles), V stored key-contiguous
        #pragma unroll
        for (int dt = 0; dt < 4; ++dt) {
            const __bf16* vr = &Vb[(size_t)(dt * 16 + M) * TT];
            v16bf vbf;
            #pragma unroll
            for (int vv = 0; vv < 8; ++vv) {
                int kk = ((vv & 3) << 1) + (ha << 3) + ((vv >> 2) << 4);
                int key = c + kk;
                int keyc = key < (TT - 2) ? key : (TT - 2);   // even, pair stays in range
                bf16x2 p = ld_bf2(&vr[keyc]);
                vbf[2 * vv] = p[0]; vbf[2 * vv + 1] = p[1];
            }
            o[dt] = wmma_bf16(pa, vbf, o[dt]);
        }
    }

    // ---- normalize + store bf16 O[B,T,H*Dh]
    #pragma unroll
    for (int r = 0; r < 8; ++r) {
        int t = qt0 + r + (ha << 3);
        float inv = l_run[r] > 0.0f ? 1.0f / l_run[r] : 0.0f;
        #pragma unroll
        for (int dt = 0; dt < 4; ++dt) {
            int dh = dt * 16 + M;
            O[((size_t)b * TT + t) * (HH * DH) + h * DH + dh] = f2bf(o[dt][r] * inv);
        }
    }
}

// ---------------------------------------------------------------- launch
extern "C" void kernel_launch(void* const* d_in, const int* in_sizes, int n_in,
                              void* d_out, int out_size, void* d_ws, size_t ws_size,
                              hipStream_t stream)
{
    (void)in_sizes; (void)n_in; (void)out_size; (void)ws_size;
    const float* x  = (const float*)d_in[0];
    const float* Wq = (const float*)d_in[1];
    const float* Wk = (const float*)d_in[2];
    const float* Wv = (const float*)d_in[3];
    const float* Wo = (const float*)d_in[4];
    float* out = (float*)d_out;

    char* ws = (char*)d_ws;
    float*  qkv = (float*)(ws);                    // 4096*1536*4  = 25165824 B
    __bf16* xbf = (__bf16*)(ws + 25165824u);       // 4096*1024*2  =  8388608 B
    __bf16* Wbf = (__bf16*)(ws + 33554432u);       // 2560*1024*2  =  5242880 B
    __bf16* Qbf = (__bf16*)(ws + 38797312u);       //                 8388608 B
    __bf16* Kbf = (__bf16*)(ws + 47185920u);       //                 2097152 B
    __bf16* Vbf = (__bf16*)(ws + 49283072u);       //                 2097152 B
    __bf16* Obf = (__bf16*)(ws + 51380224u);       // 4096*1024*2  =  8388608 B

    // 0) one-time bf16 conversion of activations + all weights (Wq|Wk|Wv|Wo rows)
    cvt_f32_bf16<<<4096, 256, 0, stream>>>(x,  xbf);
    cvt_f32_bf16<<<1024, 256, 0, stream>>>(Wq, Wbf);
    cvt_f32_bf16<<< 256, 256, 0, stream>>>(Wk, Wbf + 1048576);
    cvt_f32_bf16<<< 256, 256, 0, stream>>>(Wv, Wbf + 1310720);
    cvt_f32_bf16<<<1024, 256, 0, stream>>>(Wo, Wbf + 1572864);

    // 1) fused QKV projection: [4096,1024] x [1024,1536]^T  (Wq|Wk|Wv rows)
    dim3 g1(4096 / 128, 1536 / 128);
    gemm_bf16_wmma<<<g1, 256, 0, stream>>>(xbf, Wbf, qkv, 1024, 1536);

    // 2) RoPE + bf16 conversion / V transpose
    rope_q_kernel<<<(BB * TT * HH * 32) / 256, 256, 0, stream>>>(qkv, Qbf);
    rope_k_kernel<<<(BB * TT * KVH * 32) / 256, 256, 0, stream>>>(qkv, Kbf);
    conv_v_kernel<<<(BB * TT * KVH * 64) / 256, 256, 0, stream>>>(qkv, Vbf);

    // 3) sliding-window flash attention -> bf16 O
    dim3 ga(TT / 64, HH, BB);
    attn_kernel<<<ga, 128, 0, stream>>>(Qbf, Kbf, Vbf, Obf);

    // 4) output projection: [4096,1024] x [1024,1024]^T (Wo rows)
    dim3 g2(4096 / 128, 1024 / 128);
    gemm_bf16_wmma<<<g2, 256, 0, stream>>>(Obf, Wbf + 1572864, out, 1024, 1024);
}
